// ALCDEFTemporalGNN_42477226557518
// MI455X (gfx1250) — compile-verified, matrix operand order checked by reading, not run on hardware
//
#include <hip/hip_runtime.h>

// ---------------- problem constants (from reference) ----------------
#define N_NODES 50000
#define E_EDGES 800000
#define F_IN    16
#define H       128
#define LAYERS  3
#define G_GRAPHS 64
#define LN_EPS  1e-5f
#define NTILES  (N_NODES / 16)   // 3125 row tiles of 16
#define EPB     16               // edges per scatter block
#define WPACK   (8 * 4 * 32 * 16) // packed f16 elements per 128x128 weight

typedef _Float16 f16;
typedef __attribute__((ext_vector_type(8)))  _Float16 half8;
typedef __attribute__((ext_vector_type(16))) _Float16 half16;
typedef __attribute__((ext_vector_type(8)))  float    float8;

static __device__ __forceinline__ half16 join16(half8 lo, half8 hi) {
  half16 r;
#pragma unroll
  for (int i = 0; i < 8; ++i) { r[i] = lo[i]; r[i + 8] = hi[i]; }
  return r;
}

static __device__ __forceinline__ float8 wmma16(half16 a, half16 b, float8 c) {
  // D = A(16x32 f16) x B(32x16 f16) + C(16x16 f32)
  return __builtin_amdgcn_wmma_f32_16x16x32_f16(false, a, false, b, (short)0, c,
                                                false, false);
}

// ---------------- setup kernels ----------------
__global__ void k_init(float* deg, float* pooled, float* counts) {
  int i = blockIdx.x * blockDim.x + threadIdx.x;
  if (i < N_NODES) deg[i] = 1.0f;           // self-loop contribution
  if (i < G_GRAPHS * H) pooled[i] = 0.0f;
  if (i < G_GRAPHS) counts[i] = 0.0f;
}

__global__ void k_deg(const int* __restrict__ eidx, float* deg) {
  int e = blockIdx.x * blockDim.x + threadIdx.x;
  if (e < E_EDGES) atomicAdd(&deg[eidx[E_EDGES + e]], 1.0f);  // dst degrees
}

__global__ void k_counts(const int* __restrict__ batch, float* counts) {
  int i = blockIdx.x * blockDim.x + threadIdx.x;
  if (i < N_NODES) atomicAdd(&counts[batch[i]], 1.0f);
}

__global__ void k_dis(const float* __restrict__ deg, float* __restrict__ dis) {
  int i = blockIdx.x * blockDim.x + threadIdx.x;
  if (i < N_NODES) dis[i] = rsqrtf(fmaxf(deg[i], 1.0f));
}

// Pre-pack a row-major f32 weight [K x 128] into per-lane B-fragment layout:
// out[((ntile*KC+kc)*32 + lane)*16 + j], where halves j map to the ISA's
// 16-bit B-matrix 32x16 striping (lanes 0-15: K {0..7,16..23}; 16-31: +8).
// Each lane's 16 halves are 32B contiguous -> 2x b128 per fragment.
__global__ void k_pack_b(const float* __restrict__ W, int K, int KC,
                         f16* __restrict__ out) {
  int idx = blockIdx.x * blockDim.x + threadIdx.x;
  int total = 8 * KC * 32 * 16;
  if (idx >= total) return;
  int j    = idx & 15;
  int lane = (idx >> 4) & 31;
  int t    = idx >> 9;
  int kc = t % KC, ntile = t / KC;
  int col = ntile * 16 + (lane & 15);
  int k = kc * 32 + (j & 7) + ((j >> 3) * 16) + ((lane >> 4) * 8);
  out[idx] = (f16)((k < K) ? W[(size_t)k * H + col] : 0.0f);
}

// ---------------- encoder: h = relu(X@W1+b1)@W2+b2, one wave per 16-row tile
__global__ void __launch_bounds__(128) k_encoder(
    const float* __restrict__ X, const f16* __restrict__ w1p,
    const float* __restrict__ b1, const f16* __restrict__ w2p,
    const float* __restrict__ b2, float* __restrict__ h,
    f16* __restrict__ hf) {
  __shared__ __align__(16) f16 ldsT[4][16 * H];   // per-wave 16x128 f16 staging
  int lane = threadIdx.x & 31, wid = threadIdx.x >> 5;
  int tile = blockIdx.x * 4 + wid;
  if (tile >= NTILES) return;
  int r0 = tile * 16;
  int lr = lane & 15, hs = lane >> 4;   // hs: high-half-of-wave selector

  // A1 fragment from X (K=16 zero-padded to 32), ISA 16-bit A 16x32 layout
  half16 a1;
  {
    const float* p = X + (size_t)(r0 + lr) * F_IN + hs * 8;
    float4 x0 = *(const float4*)p;
    float4 x1 = *(const float4*)(p + 4);
    a1[0] = (f16)x0.x; a1[1] = (f16)x0.y; a1[2] = (f16)x0.z; a1[3] = (f16)x0.w;
    a1[4] = (f16)x1.x; a1[5] = (f16)x1.y; a1[6] = (f16)x1.z; a1[7] = (f16)x1.w;
#pragma unroll
    for (int i = 8; i < 16; ++i) a1[i] = (f16)0.0f;   // padded K
  }

  // layer 1: T = relu(X@W1 + b1), staged to LDS row-major f16
#pragma unroll
  for (int nt = 0; nt < 8; ++nt) {
    const f16* bp = w1p + ((size_t)nt * 32 + lane) * 16;
    half16 b = join16(*(const half8*)bp, *(const half8*)(bp + 8));
    float bias = b1[nt * 16 + lr];
    float8 c;
#pragma unroll
    for (int v = 0; v < 8; ++v) c[v] = bias;
    c = wmma16(a1, b, c);
#pragma unroll
    for (int v = 0; v < 8; ++v) {
      float y = fmaxf(c[v], 0.0f);
      ldsT[wid][(v + 8 * hs) * H + nt * 16 + lr] = (f16)y;   // C-layout -> row-major
    }
  }
  // same-wave LDS RAW: DS ops are in-order per wave; fence the compiler + DScnt
  asm volatile("s_wait_dscnt 0" ::: "memory");

  // layer 2: h = T @ W2 + b2
  half16 a2[4];
#pragma unroll
  for (int kc = 0; kc < 4; ++kc) {
    const f16* p = &ldsT[wid][lr * H + kc * 32 + hs * 8];
    a2[kc] = join16(*(const half8*)p, *(const half8*)(p + 16));
  }
#pragma unroll
  for (int nt = 0; nt < 8; ++nt) {
    float bias = b2[nt * 16 + lr];
    float8 c;
#pragma unroll
    for (int v = 0; v < 8; ++v) c[v] = bias;
#pragma unroll
    for (int kc = 0; kc < 4; ++kc) {
      const f16* bp = w2p + (((size_t)nt * 4 + kc) * 32 + lane) * 16;
      half16 b = join16(*(const half8*)bp, *(const half8*)(bp + 8));
      c = wmma16(a2[kc], b, c);
    }
#pragma unroll
    for (int v = 0; v < 8; ++v) {
      size_t row = r0 + v + 8 * hs;
      int col = nt * 16 + lr;
      h[row * H + col]  = c[v];
      hf[row * H + col] = (f16)c[v];
    }
  }
}

// ---------------- conv GEMM with folded GCN normalization ----------------
// Weights staged once per 8-wave workgroup in LDS (32KB) and shared; each
// wave computes one 16-row tile:
//   m' = (h @ W) * dis[row];  agg = conv_b + m' * dis[row]  (self-loop term)
__global__ void __launch_bounds__(256) k_conv_gemm(
    const f16* __restrict__ hf, const f16* __restrict__ wp,
    const float* __restrict__ dis, const float* __restrict__ cb,
    float* __restrict__ m, float* __restrict__ agg) {
  __shared__ __align__(16) f16 ldsW[WPACK];   // 32KB packed B fragments
  int tid = threadIdx.x;
  // cooperative stage: WPACK halves = 2048 x 16B chunks, 256 threads x 8
  {
    const float4* src = (const float4*)wp;
    float4* dst = (float4*)ldsW;
#pragma unroll
    for (int i = 0; i < 8; ++i) dst[tid + i * 256] = src[tid + i * 256];
  }
  __syncthreads();

  int lane = tid & 31, wid = tid >> 5;
  int tile = blockIdx.x * 8 + wid;
  if (tile >= NTILES) return;
  int r0 = tile * 16;
  int lr = lane & 15, hs = lane >> 4;

  half16 a[4];
#pragma unroll
  for (int kc = 0; kc < 4; ++kc) {
    const f16* p = hf + (size_t)(r0 + lr) * H + kc * 32 + hs * 8;
    a[kc] = join16(*(const half8*)p, *(const half8*)(p + 16));
  }
  float dv[8];
#pragma unroll
  for (int v = 0; v < 8; ++v) dv[v] = dis[r0 + v + 8 * hs];

#pragma unroll
  for (int nt = 0; nt < 8; ++nt) {
    float8 c = {};
#pragma unroll
    for (int kc = 0; kc < 4; ++kc) {
      const f16* bp = &ldsW[(((size_t)nt * 4 + kc) * 32 + lane) * 16];
      half16 b = join16(*(const half8*)bp, *(const half8*)(bp + 8));
      c = wmma16(a[kc], b, c);
    }
    int col = nt * 16 + lr;
    float bias = cb[col];
#pragma unroll
    for (int v = 0; v < 8; ++v) {
      size_t row = r0 + v + 8 * hs;
      float val = c[v] * dv[v];
      m[row * H + col]   = val;
      agg[row * H + col] = bias + val * dv[v];
    }
  }
}

// ---------------- edge scatter: agg[dst] += m'[src] * dis[dst] ----------------
__global__ void __launch_bounds__(128) k_scatter(
    const int* __restrict__ eidx, const float* __restrict__ m,
    const float* __restrict__ dis, float* __restrict__ agg) {
  int tid = threadIdx.x;          // channel 0..127
  int e0 = blockIdx.x * EPB;
#pragma unroll 1
  for (int k = 0; k < EPB; ++k) {
    int e = e0 + k;
    if (e >= E_EDGES) return;
    int s = eidx[e];
    int d = eidx[E_EDGES + e];
    if (k + 1 < EPB && e + 1 < E_EDGES)
      __builtin_prefetch(&m[(size_t)eidx[e + 1] * H], 0, 0);
    float val = m[(size_t)s * H + tid] * dis[d];
    atomicAdd(&agg[(size_t)d * H + tid], val);
  }
}

// ---------------- LayerNorm + ReLU + residual (+ optional pooling) ----------
__global__ void __launch_bounds__(128) k_ln_relu(
    const float* __restrict__ agg, const float* __restrict__ g,
    const float* __restrict__ b, float* __restrict__ h, f16* __restrict__ hf,
    const int* __restrict__ batch, float* __restrict__ pooled,
    int has_res, int do_pool) {
  int lane = threadIdx.x & 31, wid = threadIdx.x >> 5;
  int node = blockIdx.x * 4 + wid;
  if (node >= N_NODES) return;
  float x[4], s = 0.f, s2 = 0.f;
#pragma unroll
  for (int j = 0; j < 4; ++j) {
    x[j] = agg[(size_t)node * H + lane + 32 * j];
    s += x[j]; s2 += x[j] * x[j];
  }
#pragma unroll
  for (int off = 16; off > 0; off >>= 1) {   // wave32 reduction
    s  += __shfl_xor(s, off, 32);
    s2 += __shfl_xor(s2, off, 32);
  }
  float mean = s * (1.0f / H);
  float var  = s2 * (1.0f / H) - mean * mean;
  float rs = rsqrtf(var + LN_EPS);
  int bg = do_pool ? batch[node] : 0;
#pragma unroll
  for (int j = 0; j < 4; ++j) {
    int c = lane + 32 * j;
    float y = (x[j] - mean) * rs * g[c] + b[c];
    y = fmaxf(y, 0.0f);
    if (has_res) y += h[(size_t)node * H + c];
    h[(size_t)node * H + c]  = y;
    hf[(size_t)node * H + c] = (f16)y;
    if (do_pool) atomicAdd(&pooled[bg * H + c], y);
  }
}

// ---------------- head: pool-normalize + MLP 128->64->32->1 + softplus -------
__global__ void __launch_bounds__(256) k_head(
    const float* __restrict__ pooled, const float* __restrict__ counts,
    const float* __restrict__ W1, const float* __restrict__ b1,
    const float* __restrict__ W2, const float* __restrict__ b2,
    const float* __restrict__ W3, const float* __restrict__ b3,
    float* __restrict__ out) {
  __shared__ float P[G_GRAPHS * H];
  __shared__ float Z1[G_GRAPHS * 64];
  __shared__ float Z2[G_GRAPHS * 32];
  int tid = threadIdx.x;
  for (int i = tid; i < G_GRAPHS * H; i += 256)
    P[i] = pooled[i] / fmaxf(counts[i / H], 1.0f);
  __syncthreads();
  for (int i = tid; i < G_GRAPHS * 64; i += 256) {
    int gi = i / 64, o = i % 64;
    float acc = b1[o];
    for (int k = 0; k < H; ++k) acc += P[gi * H + k] * W1[k * 64 + o];
    Z1[i] = fmaxf(acc, 0.0f);
  }
  __syncthreads();
  for (int i = tid; i < G_GRAPHS * 32; i += 256) {
    int gi = i / 32, o = i % 32;
    float acc = b2[o];
    for (int k = 0; k < 64; ++k) acc += Z1[gi * 64 + k] * W2[k * 32 + o];
    Z2[i] = fmaxf(acc, 0.0f);
  }
  __syncthreads();
  if (tid < G_GRAPHS) {
    float acc = b3[0];
    for (int k = 0; k < 32; ++k) acc += Z2[tid * 32 + k] * W3[k];
    out[tid] = (acc > 20.0f) ? acc : log1pf(expf(acc));
  }
}

// ---------------- launcher ----------------
extern "C" void kernel_launch(void* const* d_in, const int* in_sizes, int n_in,
                              void* d_out, int out_size, void* d_ws,
                              size_t ws_size, hipStream_t stream) {
  (void)in_sizes; (void)n_in; (void)out_size; (void)ws_size;
  const float* X     = (const float*)d_in[0];
  const int*   eidx  = (const int*)d_in[1];
  const int*   batch = (const int*)d_in[2];
  const float* encW1 = (const float*)d_in[3];
  const float* encb1 = (const float*)d_in[4];
  const float* encW2 = (const float*)d_in[5];
  const float* encb2 = (const float*)d_in[6];
  const float* convW = (const float*)d_in[7];
  const float* convb = (const float*)d_in[8];
  const float* lng   = (const float*)d_in[9];
  const float* lnb   = (const float*)d_in[10];
  const float* hW1   = (const float*)d_in[11];
  const float* hb1   = (const float*)d_in[12];
  const float* hW2   = (const float*)d_in[13];
  const float* hb2   = (const float*)d_in[14];
  const float* hW3   = (const float*)d_in[15];
  const float* hb3   = (const float*)d_in[16];
  float* out = (float*)d_out;

  char* ws = (char*)d_ws;
  size_t off = 0;
  auto alloc = [&](size_t bytes) -> void* {
    void* p = ws + off;
    off = (off + bytes + 255) & ~(size_t)255;
    return p;
  };
  float* deg    = (float*)alloc((size_t)N_NODES * 4);
  float* dis    = (float*)alloc((size_t)N_NODES * 4);
  float* h      = (float*)alloc((size_t)N_NODES * H * 4);
  f16*   hf     = (f16*)  alloc((size_t)N_NODES * H * 2);
  float* m      = (float*)alloc((size_t)N_NODES * H * 4);
  float* agg    = (float*)alloc((size_t)N_NODES * H * 4);
  float* pooled = (float*)alloc((size_t)G_GRAPHS * H * 4);
  float* counts = (float*)alloc((size_t)G_GRAPHS * 4);
  f16*   w1p    = (f16*)alloc((size_t)8 * 1 * 32 * 16 * 2);
  f16*   w2p    = (f16*)alloc((size_t)WPACK * 2);
  f16*   wcp    = (f16*)alloc((size_t)LAYERS * WPACK * 2);

  int nbN = (N_NODES + 255) / 256;
  k_init  <<<nbN, 256, 0, stream>>>(deg, pooled, counts);
  k_deg   <<<(E_EDGES + 255) / 256, 256, 0, stream>>>(eidx, deg);
  k_counts<<<nbN, 256, 0, stream>>>(batch, counts);
  k_dis   <<<nbN, 256, 0, stream>>>(deg, dis);

  k_pack_b<<<(8 * 1 * 32 * 16 + 255) / 256, 256, 0, stream>>>(encW1, F_IN, 1, w1p);
  k_pack_b<<<(WPACK + 255) / 256, 256, 0, stream>>>(encW2, H, 4, w2p);
  for (int l = 0; l < LAYERS; ++l)
    k_pack_b<<<(WPACK + 255) / 256, 256, 0, stream>>>(
        convW + (size_t)l * H * H, H, 4, wcp + (size_t)l * WPACK);

  k_encoder<<<(NTILES + 3) / 4, 128, 0, stream>>>(X, w1p, encb1, w2p, encb2, h, hf);

  for (int l = 0; l < LAYERS; ++l) {
    k_conv_gemm<<<(NTILES + 7) / 8, 256, 0, stream>>>(
        hf, wcp + (size_t)l * WPACK, dis, convb + (size_t)l * H, m, agg);
    k_scatter<<<(E_EDGES + EPB - 1) / EPB, 128, 0, stream>>>(eidx, m, dis, agg);
    k_ln_relu<<<(N_NODES + 3) / 4, 128, 0, stream>>>(
        agg, lng + (size_t)l * H, lnb + (size_t)l * H, h, hf, batch, pooled,
        l > 0 ? 1 : 0, l == LAYERS - 1 ? 1 : 0);
  }
  k_head<<<1, 256, 0, stream>>>(pooled, counts, hW1, hb1, hW2, hb2, hW3, hb3, out);
}
